// DeepSeekMLP_88605175316748
// MI455X (gfx1250) — 8-bit matrix operand order repaired and checked by reading, compiled, not run on hardware
//
#include <hip/hip_runtime.h>
#include <stdint.h>

typedef __attribute__((ext_vector_type(16))) int   v16i;
typedef __attribute__((ext_vector_type(8)))  float v8f;

#define FP8_MAX 448.0f

__device__ __forceinline__ unsigned char f32_to_e4m3_sw(float x) {
  unsigned char s = (__float_as_uint(x) >> 31) ? 0x80 : 0x00;
  float a = fabsf(x);
  if (!(a > 0.0f)) return s;
  if (a >= 464.0f) return (unsigned char)(s | 0x7E);
  int e; float mf = frexpf(a, &e);
  int E = e + 6;
  if (E >= 1) {
    int q = (int)rintf(mf * 16.0f);
    if (q == 16) { q = 8; ++E; }
    if (E > 15) return (unsigned char)(s | 0x7E);
    return (unsigned char)(s | (E << 3) | (q & 7));
  }
  int q = (int)rintf(a * 512.0f);
  if (q > 7) return (unsigned char)(s | 0x08);
  return (unsigned char)(s | q);
}

__device__ __forceinline__ unsigned pack4_e4m3(float a, float b, float c, float d) {
#if __has_builtin(__builtin_amdgcn_cvt_pk_fp8_f32)
  int p = 0;
  p = __builtin_amdgcn_cvt_pk_fp8_f32(a, b, p, false);
  p = __builtin_amdgcn_cvt_pk_fp8_f32(c, d, p, true);
  return (unsigned)p;
#else
  return (unsigned)f32_to_e4m3_sw(a) |
         ((unsigned)f32_to_e4m3_sw(b) << 8) |
         ((unsigned)f32_to_e4m3_sw(c) << 16) |
         ((unsigned)f32_to_e4m3_sw(d) << 24);
#endif
}

__device__ __forceinline__ float wave_amax(float v) {
#pragma unroll
  for (int off = 16; off >= 1; off >>= 1)
    v = fmaxf(v, __shfl_xor(v, off, 32));
  return v;
}

__global__ void quant_rows_128(const float* __restrict__ x, uint8_t* __restrict__ q,
                               float* __restrict__ s, int K, int KB, int M) {
  int g    = blockIdx.x * 8 + (threadIdx.x >> 5);
  int lane = threadIdx.x & 31;
  int row  = g / KB, kb = g % KB;
  const float4 v = *(const float4*)(x + (size_t)row * K + (size_t)kb * 128 + lane * 4);
  float amax = fmaxf(fmaxf(fabsf(v.x), fabsf(v.y)), fmaxf(fabsf(v.z), fabsf(v.w)));
  amax = wave_amax(amax);
  float scale = fmaxf(amax, 1e-4f) * (1.0f / FP8_MAX);
  float r = 1.0f / scale;
  *(unsigned*)(q + (size_t)row * K + (size_t)kb * 128 + lane * 4) =
      pack4_e4m3(v.x * r, v.y * r, v.z * r, v.w * r);
  if (lane == 0) s[(size_t)kb * M + row] = scale;
}

__global__ void quant_wgt_128x128(const float* __restrict__ w, uint8_t* __restrict__ q,
                                  float* __restrict__ s, int K, int KB) {
  int nb = blockIdx.x / KB, kb = blockIdx.x % KB;
  int tid = threadIdx.x;
  __shared__ float red[256];
  const float* wt = w + (size_t)nb * 128 * K + (size_t)kb * 128;
  float amax = 0.0f;
  for (int i = tid; i < 128 * 128 / 4; i += 256) {
    int row  = i >> 5;
    int col4 = (i & 31) * 4;
    float4 v = *(const float4*)(wt + (size_t)row * K + col4);
    amax = fmaxf(amax, fmaxf(fmaxf(fabsf(v.x), fabsf(v.y)),
                             fmaxf(fabsf(v.z), fabsf(v.w))));
  }
  red[tid] = amax;
  __syncthreads();
  for (int stp = 128; stp > 0; stp >>= 1) {
    if (tid < stp) red[tid] = fmaxf(red[tid], red[tid + stp]);
    __syncthreads();
  }
  float scale = fmaxf(red[0], 1e-4f) * (1.0f / FP8_MAX);
  float r = 1.0f / scale;
  uint8_t* qt = q + (size_t)nb * 128 * K + (size_t)kb * 128;
  for (int i = tid; i < 128 * 128 / 4; i += 256) {
    int row  = i >> 5;
    int col4 = (i & 31) * 4;
    float4 v = *(const float4*)(wt + (size_t)row * K + col4);
    *(unsigned*)(qt + (size_t)row * K + col4) =
        pack4_e4m3(v.x * r, v.y * r, v.z * r, v.w * r);
  }
  if (tid == 0) s[(size_t)nb * KB + kb] = scale;
}

__global__ void swiglu_quant(const float* __restrict__ h, uint8_t* __restrict__ q,
                             float* __restrict__ s, int E, int GB, int M) {
  int g    = blockIdx.x * 8 + (threadIdx.x >> 5);
  int lane = threadIdx.x & 31;
  int row  = g / GB, gb = g % GB;
  const float* hr = h + (size_t)row * (2 * E);
  float4 gt = *(const float4*)(hr + (size_t)gb * 128 + lane * 4);
  float4 up = *(const float4*)(hr + E + (size_t)gb * 128 + lane * 4);
  float4 a;
  a.x = gt.x * (1.0f / (1.0f + __expf(-gt.x))) * up.x;
  a.y = gt.y * (1.0f / (1.0f + __expf(-gt.y))) * up.y;
  a.z = gt.z * (1.0f / (1.0f + __expf(-gt.z))) * up.z;
  a.w = gt.w * (1.0f / (1.0f + __expf(-gt.w))) * up.w;
  float amax = fmaxf(fmaxf(fabsf(a.x), fabsf(a.y)), fmaxf(fabsf(a.z), fabsf(a.w)));
  amax = wave_amax(amax);
  float scale = fmaxf(amax, 1e-4f) * (1.0f / FP8_MAX);
  float r = 1.0f / scale;
  *(unsigned*)(q + (size_t)row * E + (size_t)gb * 128 + lane * 4) =
      pack4_e4m3(a.x * r, a.y * r, a.z * r, a.w * r);
  if (lane == 0) s[(size_t)gb * M + row] = scale;
}

__global__ __launch_bounds__(128) void fp8_gemm_nt_scaled(
    const uint8_t* __restrict__ qA, const float* __restrict__ sAt,
    const uint8_t* __restrict__ qB, const float* __restrict__ sB,
    float* __restrict__ out, int M, int N, int K) {
  const int KB   = K >> 7;
  const int lane = threadIdx.x & 31;
  const int wave = threadIdx.x >> 5;
  const int l15  = lane & 15;
  const int hi   = lane >> 4;
  const int base = hi << 3;
  const int col0 = (blockIdx.x * 4 + wave) * 64;
  const int row0 = blockIdx.y * 16;

  const uint8_t* ap = qA + (size_t)(row0 + l15) * K + (hi << 4);
  const uint8_t* bp = qB + (size_t)(col0 + l15) * K + (hi << 4);
  const float* sBblk = sB + (size_t)(col0 >> 7) * KB;
  const float* sA0   = sAt + row0 + base;

  v8f acc[4];
#pragma unroll
  for (int t = 0; t < 4; ++t) acc[t] = (v8f){0.f,0.f,0.f,0.f,0.f,0.f,0.f,0.f};
  const v8f zero = {0.f,0.f,0.f,0.f,0.f,0.f,0.f,0.f};

  v8f pend = zero;
  float scc[8], scp[8];
#pragma unroll
  for (int i = 0; i < 8; ++i) scc[i] = 0.0f;

  for (int kb = 0; kb < KB; ++kb) {
    const size_t ko = (size_t)kb << 7;
    v16i a;
#pragma unroll
    for (int j = 0; j < 4; ++j) {
      uint4 d = *(const uint4*)(ap + ko + 32 * j);
      a[4 * j + 0] = (int)d.x; a[4 * j + 1] = (int)d.y;
      a[4 * j + 2] = (int)d.z; a[4 * j + 3] = (int)d.w;
    }
#pragma unroll
    for (int i = 0; i < 8; ++i) scp[i] = scc[i];
    float4 s0 = *(const float4*)(sA0 + (size_t)kb * M);
    float4 s1 = *(const float4*)(sA0 + (size_t)kb * M + 4);
    float  sb = sBblk[kb];
    scc[0] = s0.x * sb; scc[1] = s0.y * sb; scc[2] = s0.z * sb; scc[3] = s0.w * sb;
    scc[4] = s1.x * sb; scc[5] = s1.y * sb; scc[6] = s1.z * sb; scc[7] = s1.w * sb;

    __builtin_prefetch(ap + ko + 128, 0, 1);
    __builtin_prefetch(bp + ko + 128, 0, 1);

#pragma unroll
    for (int t = 0; t < 4; ++t) {
      const uint8_t* b8 = bp + (size_t)(16 * t) * K + ko;
      v16i b;
#pragma unroll
      for (int gq = 0; gq < 4; ++gq) {
        uint4 d = *(const uint4*)(b8 + 32 * gq);
        b[4 * gq + 0] = (int)d.x; b[4 * gq + 1] = (int)d.y;
        b[4 * gq + 2] = (int)d.z; b[4 * gq + 3] = (int)d.w;
      }
      v8f p = __builtin_amdgcn_wmma_f32_16x16x128_fp8_fp8(a, b, (short)0, zero,
                                                          false, false);
      const int tp = (t + 3) & 3;
#pragma unroll
      for (int rr = 0; rr < 8; ++rr) {
        float sc = (t == 0) ? scp[rr] : scc[rr];
        acc[tp][rr] = fmaf(pend[rr], sc, acc[tp][rr]);
      }
      pend = p;
    }
  }
#pragma unroll
  for (int rr = 0; rr < 8; ++rr)
    acc[3][rr] = fmaf(pend[rr], scc[rr], acc[3][rr]);

  float* orow = out + (size_t)(row0 + base) * N + col0 + l15;
#pragma unroll
  for (int t = 0; t < 4; ++t)
#pragma unroll
    for (int rr = 0; rr < 8; ++rr)
      orow[(size_t)rr * N + 16 * t] = acc[t][rr];
}

extern "C" void kernel_launch(void* const* d_in, const int* in_sizes, int n_in,
                              void* d_out, int out_size, void* d_ws, size_t ws_size,
                              hipStream_t stream) {
  const float* x   = (const float*)d_in[0];
  const float* wug = (const float*)d_in[1];
  const float* wdn = (const float*)d_in[2];
  float* out = (float*)d_out;
  (void)in_sizes; (void)n_in; (void)out_size; (void)ws_size;

  const int M = 1024, H = 7168, E = 2048, N1 = 2 * E;
  const int KB1 = H / 128;
  const int GB  = E / 128;
  const int NB1 = N1 / 128;
  const int NB2 = H / 128;

  uint8_t* ws = (uint8_t*)d_ws;
  size_t off = 0;
  auto take = [&](size_t bytes) -> uint8_t* {
    uint8_t* p = ws + off;
    off += (bytes + 255) & ~(size_t)255;
    return p;
  };
  uint8_t* qx   = take((size_t)M * H);
  float*   sx   = (float*)take((size_t)KB1 * M * 4);
  uint8_t* qwug = take((size_t)N1 * H);
  float*   swug = (float*)take((size_t)NB1 * KB1 * 4);
  float*   hug  = (float*)take((size_t)M * N1 * 4);
  uint8_t* qact = take((size_t)M * E);
  float*   sact = (float*)take((size_t)GB * M * 4);
  uint8_t* qwdn = take((size_t)H * E);
  float*   swdn = (float*)take((size_t)NB2 * GB * 4);

  quant_rows_128<<<dim3((M * KB1) / 8), 256, 0, stream>>>(x, qx, sx, H, KB1, M);
  quant_wgt_128x128<<<dim3(NB1 * KB1), 256, 0, stream>>>(wug, qwug, swug, H, KB1);
  fp8_gemm_nt_scaled<<<dim3(N1 / 256, M / 16), 128, 0, stream>>>(qx, sx, qwug, swug,
                                                                 hug, M, N1, H);
  swiglu_quant<<<dim3((M * GB) / 8), 256, 0, stream>>>(hug, qact, sact, E, GB, M);
  quant_wgt_128x128<<<dim3(NB2 * GB), 256, 0, stream>>>(wdn, qwdn, swdn, E, GB);
  fp8_gemm_nt_scaled<<<dim3(H / 256, M / 16), 128, 0, stream>>>(qact, sact, qwdn, swdn,
                                                                out, M, H, E);
}
